// SpatioTemporalAttention_22651657519414
// MI455X (gfx1250) — matrix operand order repaired and checked by reading, not compiled, not run on hardware
//
#include <hip/hip_runtime.h>

typedef __attribute__((ext_vector_type(16))) _Float16 v16h;
typedef __attribute__((ext_vector_type(8)))  float    v8f;

#define TV 12800          // T*V = 256*50
#define NT 256
#define NV 50

// ---- WMMA helpers (gfx1250 wave32, v_wmma_f32_16x16x32_f16) ----------------

static __device__ __forceinline__ v8f wmma_f16(v16h a, v16h b, v8f c) {
  return __builtin_amdgcn_wmma_f32_16x16x32_f16(
      /*neg_a=*/false, a, /*neg_b=*/false, b,
      /*c_mod=*/(short)0, c, /*reuse_a=*/false, /*reuse_b=*/false);
}

// A fragment: A is row-major [m][k] in LDS with row pitch `pitch` halves.
// ISA layout: lane (m = l&15, hi = l>>4), element j -> K = hi*8 + (j&7) + (j>=8 ? 16 : 0)
static __device__ __forceinline__ v16h load_a_frag(const _Float16* A, int pitch,
                                                   int mb, int k0, int lane) {
  const int m = lane & 15, hi = lane >> 4;
  const _Float16* p = A + (mb + m) * pitch + k0 + hi * 8;
  v16h a;
#pragma unroll
  for (int j = 0; j < 8; ++j) { a[j] = p[j]; a[8 + j] = p[16 + j]; }
  return a;
}

// B fragment: BT is column-major-transposed [col][k] in LDS, row pitch `pitch`.
// ISA layout: lane (n = l&15, hi = l>>4), element j -> K = hi*8 + (j&7) + (j>=8 ? 16 : 0)
static __device__ __forceinline__ v16h load_b_frag(const _Float16* BT, int pitch,
                                                   int cb, int k0, int lane) {
  const int nn = lane & 15, hi = lane >> 4;
  const _Float16* p = BT + (cb + nn) * pitch + k0 + hi * 8;
  v16h b;
#pragma unroll
  for (int j = 0; j < 8; ++j) { b[j] = p[j]; b[8 + j] = p[16 + j]; }
  return b;
}

// ---- Kernel A: fused q/k/v 1x1-conv projections, f32 in -> f16 out ---------
// y[n,o,tv] = sum_c W[o,c] x[n,c,tv] + b[o]; per block: one n, 64 tv columns.
__global__ __launch_bounds__(256) void qkv_proj_kernel(
    const float* __restrict__ xq, const float* __restrict__ xkv,
    const float* __restrict__ qw, const float* __restrict__ qb,
    const float* __restrict__ kw, const float* __restrict__ kb,
    const float* __restrict__ vw, const float* __restrict__ vb,
    _Float16* __restrict__ qo, _Float16* __restrict__ ko, _Float16* __restrict__ vo)
{
  __shared__ __attribute__((aligned(16))) _Float16 sW[3 * 64 * 72];   // A: [o][c]
  __shared__ __attribute__((aligned(16))) _Float16 sXq[64 * 72];      // BT: [col][c]
  __shared__ __attribute__((aligned(16))) _Float16 sXkv[64 * 72];
  __shared__ float sBias[3 * 64];

  const int n = blockIdx.y;
  const int cb64 = blockIdx.x * 64;
  const int tid = threadIdx.x;

  for (int idx = tid; idx < 4096; idx += 256) {
    const int o = idx >> 6, c = idx & 63;
    sW[0 * 64 * 72 + o * 72 + c] = (_Float16)qw[o * 64 + c];
    sW[1 * 64 * 72 + o * 72 + c] = (_Float16)kw[o * 64 + c];
    sW[2 * 64 * 72 + o * 72 + c] = (_Float16)vw[o * 64 + c];
  }
  if (tid < 64) {
    sBias[tid]       = qb[tid];
    sBias[64 + tid]  = kb[tid];
    sBias[128 + tid] = vb[tid];
  }
  const float* xq_n  = xq  + (size_t)n * 64 * TV + cb64;
  const float* xkv_n = xkv + (size_t)n * 64 * TV + cb64;
  for (int idx = tid; idx < 4096; idx += 256) {
    const int c = idx >> 6, col = idx & 63;                 // coalesced over col
    sXq [col * 72 + c] = (_Float16)xq_n [c * TV + col];
    sXkv[col * 72 + c] = (_Float16)xkv_n[c * TV + col];
  }
  __syncthreads();

  const int wave = tid >> 5, lane = tid & 31;
  const int nn = lane & 15, hi = lane >> 4;
#pragma unroll
  for (int i = 0; i < 6; ++i) {
    const int tile = wave + 8 * i;          // 48 tiles: 3 outputs x 4 mtiles x 4 ctiles
    const int s = tile / 16, rem = tile % 16;
    const int mt = rem & 3, ct = rem >> 2;
    const _Float16* A  = sW + s * 64 * 72;
    const _Float16* BT = (s == 0) ? sXq : sXkv;
    v8f acc = {};
    acc = wmma_f16(load_a_frag(A, 72, mt * 16, 0,  lane),
                   load_b_frag(BT, 72, ct * 16, 0,  lane), acc);
    acc = wmma_f16(load_a_frag(A, 72, mt * 16, 32, lane),
                   load_b_frag(BT, 72, ct * 16, 32, lane), acc);
    _Float16* outsel = (s == 0) ? qo : (s == 1 ? ko : vo);
    _Float16* dst = outsel + (size_t)n * 64 * TV + (cb64 + ct * 16 + nn);
#pragma unroll
    for (int r = 0; r < 8; ++r) {
      const int o = mt * 16 + r + 8 * hi;
      dst[(size_t)o * TV] = (_Float16)(acc[r] + sBias[s * 64 + o]);
    }
  }
}

// ---- Kernel B: scores + tanh + graph bias ---------------------------------
// scores[n,h,u,v] = (1/2048) sum_{c,t} q[n,h*8+c,t,u] k[n,h*8+c,t,v]
// attnT[n,h,vv,u] = tanh(scores)*alpha[h] + G[u,vv]  (f16, transposed, 64x64 padded)
__global__ __launch_bounds__(256) void attn_kernel(
    const _Float16* __restrict__ qf, const _Float16* __restrict__ kf,
    const float* __restrict__ gattn, const float* __restrict__ alphas,
    _Float16* __restrict__ attnT)
{
  __shared__ __attribute__((aligned(16))) _Float16 sQ [64 * 40];  // A: [u][tt]
  __shared__ __attribute__((aligned(16))) _Float16 sKT[64 * 40];  // BT: [vv][tt]

  const int h = blockIdx.x, n = blockIdx.y;
  const int tid = threadIdx.x, wave = tid >> 5, lane = tid & 31;

  const _Float16* qbase = qf + ((size_t)n * 64 + h * 8) * (NT * NV);
  const _Float16* kbase = kf + ((size_t)n * 64 + h * 8) * (NT * NV);

  const int tile0 = wave * 2;             // 16 output tiles, 2 per wave
  const int mt0 = tile0 >> 2,       ct0 = tile0 & 3;
  const int mt1 = (tile0 + 1) >> 2, ct1 = (tile0 + 1) & 3;
  v8f acc0 = {}, acc1 = {};

  for (int ks = 0; ks < 64; ++ks) {       // K = hc*T = 2048, chunks of 32 (c,t)
    const int c = ks >> 3, t0 = (ks & 7) * 32;
    const _Float16* qc = qbase + ((size_t)c * NT + t0) * NV;
    const _Float16* kc = kbase + ((size_t)c * NT + t0) * NV;
    if (ks + 1 < 64) {                    // prefetch next slab (global_prefetch_b8)
      const int cN = (ks + 1) >> 3, tN = ((ks + 1) & 7) * 32;
      __builtin_prefetch(qbase + ((size_t)cN * NT + tN) * NV + tid * 8, 0, 1);
      __builtin_prefetch(kbase + ((size_t)cN * NT + tN) * NV + tid * 8, 0, 1);
    }
    __syncthreads();                      // prior WMMA reads of LDS done
    for (int idx = tid; idx < 2048; idx += 256) {
      const int tt = idx >> 6, u = idx & 63;   // coalesced over u
      const _Float16 z = (_Float16)0.f;
      sQ [u * 40 + tt] = (u < NV) ? qc[tt * NV + u] : z;
      sKT[u * 40 + tt] = (u < NV) ? kc[tt * NV + u] : z;
    }
    __syncthreads();
    acc0 = wmma_f16(load_a_frag(sQ, 40, mt0 * 16, 0, lane),
                    load_b_frag(sKT, 40, ct0 * 16, 0, lane), acc0);
    acc1 = wmma_f16(load_a_frag(sQ, 40, mt1 * 16, 0, lane),
                    load_b_frag(sKT, 40, ct1 * 16, 0, lane), acc1);
  }

  const float alpha = alphas[h];
  const float inv = 1.0f / 2048.0f;
  _Float16* ob = attnT + ((size_t)n * 8 + h) * 64 * 64;
  const int nn = lane & 15, hi = lane >> 4;
#pragma unroll
  for (int p = 0; p < 2; ++p) {
    const v8f a = p ? acc1 : acc0;
    const int mt = p ? mt1 : mt0, ct = p ? ct1 : ct0;
    const int vv = ct * 16 + nn;
#pragma unroll
    for (int r = 0; r < 8; ++r) {
      const int u = mt * 16 + r + 8 * hi;
      const float g = (u < NV && vv < NV) ? gattn[u * NV + vv] : 0.f;
      ob[vv * 64 + u] = (_Float16)(tanhf(a[r] * inv) * alpha + g);
    }
  }
}

// ---- Kernel C: fused values GEMM + o-projection ---------------------------
// out[n,o,t,v] = sum_h sum_c ow[o,h*64+c] * (sum_u v[n,c,t,u] attn[n,h,u,v]) + ob[o]
// One block: (n, 4 timesteps). out tile kept in registers across the head loop.
__global__ __launch_bounds__(256) void out_proj_kernel(
    const _Float16* __restrict__ vf, const _Float16* __restrict__ attnT,
    const float* __restrict__ ow, const float* __restrict__ obias,
    float* __restrict__ out)
{
  extern __shared__ __attribute__((aligned(16))) _Float16 smem[];
  _Float16* sV     = smem;                  // A: rows (c*4+t) x k=u, pitch 72
  _Float16* sTmpB  = smem + 256 * 72;       // BT: cols (t*64+vv) x k=c, pitch 72
  _Float16* sAttnT = smem + 2 * 256 * 72;   // BT: [vv][u], pitch 72
  _Float16* sOw    = sAttnT + 64 * 72;      // A: [o][c], pitch 72

  const int tb = blockIdx.x, n = blockIdx.y;
  const int t0 = tb * 4;
  const int tid = threadIdx.x, wave = tid >> 5, lane = tid & 31;
  const int nn = lane & 15, hi = lane >> 4;

  const _Float16* vbase = vf + (size_t)n * 64 * (NT * NV);
  for (int idx = tid; idx < 256 * 64; idx += 256) {
    const int row = idx >> 6, u = idx & 63;     // row = c*4 + t
    const int c = row >> 2, t = row & 3;
    sV[row * 72 + u] =
        (u < NV) ? vbase[((size_t)c * NT + t0 + t) * NV + u] : (_Float16)0.f;
  }

  v8f oacc[8];
  const v8f vzero = {};
#pragma unroll
  for (int i = 0; i < 8; ++i) oacc[i] = vzero;

  for (int h = 0; h < 8; ++h) {
    __syncthreads();                            // prior reads of sOw/sAttnT/sV done
    const _Float16* ab = attnT + ((size_t)n * 8 + h) * 64 * 64;
    for (int idx = tid; idx < 4096; idx += 256) {
      const int r0 = idx >> 6, q = idx & 63;
      sAttnT[r0 * 72 + q] = ab[r0 * 64 + q];              // [vv][u]
      sOw[r0 * 72 + q] = (_Float16)ow[r0 * 512 + h * 64 + q];  // [o][c]
    }
    __syncthreads();

    // step 2: tmp[(c,t),vv] = sum_u v[c,t,u] attn_h[u,vv]; write transposed
#pragma unroll
    for (int i = 0; i < 8; ++i) {
      const int tile = wave + 8 * i;            // 64 tiles: 16 mtiles x 4 ctiles
      const int mt = tile >> 2, ct = tile & 3;
      v8f t2 = {};
      t2 = wmma_f16(load_a_frag(sV, 72, mt * 16, 0,  lane),
                    load_b_frag(sAttnT, 72, ct * 16, 0,  lane), t2);
      t2 = wmma_f16(load_a_frag(sV, 72, mt * 16, 32, lane),
                    load_b_frag(sAttnT, 72, ct * 16, 32, lane), t2);
      const int vv = ct * 16 + nn;
#pragma unroll
      for (int r = 0; r < 8; ++r) {
        const int rix = mt * 16 + r + 8 * hi;   // = c*4 + t
        const int c = rix >> 2, t = rix & 3;
        sTmpB[(t * 64 + vv) * 72 + c] = (_Float16)t2[r];
      }
    }
    __syncthreads();

    // step 3: out[o,(t,vv)] += ow_h[o,c] * tmp[c,(t,vv)]
#pragma unroll
    for (int i = 0; i < 8; ++i) {
      const int tile = wave + 8 * i;            // 64 tiles: 4 mtiles x 16 ctiles
      const int mt = tile >> 4, ct = tile & 15;
      oacc[i] = wmma_f16(load_a_frag(sOw, 72, mt * 16, 0,  lane),
                         load_b_frag(sTmpB, 72, ct * 16, 0,  lane), oacc[i]);
      oacc[i] = wmma_f16(load_a_frag(sOw, 72, mt * 16, 32, lane),
                         load_b_frag(sTmpB, 72, ct * 16, 32, lane), oacc[i]);
    }
  }

#pragma unroll
  for (int i = 0; i < 8; ++i) {
    const int tile = wave + 8 * i;
    const int mt = tile >> 4, ct = tile & 15;
    const int col = ct * 16 + nn;
    const int t = col >> 6, vv = col & 63;
    if (vv < NV) {
#pragma unroll
      for (int r = 0; r < 8; ++r) {
        const int o = mt * 16 + r + 8 * hi;
        out[(((size_t)n * 64 + o) * NT + (t0 + t)) * NV + vv] =
            oacc[i][r] + obias[o];
      }
    }
  }
}

// ---- host launch -----------------------------------------------------------
extern "C" void kernel_launch(void* const* d_in, const int* in_sizes, int n_in,
                              void* d_out, int out_size, void* d_ws, size_t ws_size,
                              hipStream_t stream) {
  (void)in_sizes; (void)n_in; (void)out_size; (void)ws_size;
  const float* x_q    = (const float*)d_in[0];
  const float* x_kv   = (const float*)d_in[1];
  const float* q_w    = (const float*)d_in[2];
  const float* q_b    = (const float*)d_in[3];
  const float* k_w    = (const float*)d_in[4];
  const float* k_b    = (const float*)d_in[5];
  const float* v_w    = (const float*)d_in[6];
  const float* v_b    = (const float*)d_in[7];
  const float* o_w    = (const float*)d_in[8];
  const float* o_b    = (const float*)d_in[9];
  const float* alphas = (const float*)d_in[10];
  const float* gattn  = (const float*)d_in[11];
  float* out = (float*)d_out;

  const size_t QKV = (size_t)32 * 64 * NT * NV;     // elements per projection
  _Float16* qf    = (_Float16*)d_ws;
  _Float16* kf    = qf + QKV;
  _Float16* vfp   = kf + QKV;
  _Float16* attnT = vfp + QKV;                      // 32*8*64*64 f16

  qkv_proj_kernel<<<dim3(200, 32), 256, 0, stream>>>(
      x_q, x_kv, q_w, q_b, k_w, k_b, v_w, v_b, qf, kf, vfp);
  attn_kernel<<<dim3(8, 32), 256, 0, stream>>>(qf, kf, gattn, alphas, attnT);
  const size_t ldsC = (size_t)(2 * 256 * 72 + 2 * 64 * 72) * sizeof(_Float16); // 92160 B
  out_proj_kernel<<<dim3(64, 32), 256, ldsC, stream>>>(vfp, attnT, o_w, o_b, out);
}
